// AttentionJacobianKernel_85899345920194
// MI455X (gfx1250) — matrix operand order repaired and checked by reading, not compiled, not run on hardware
//
#include <hip/hip_runtime.h>

#define BQ 16
#define QN 256
#define SN 2048
#define DN 64
#define CHUNK 64
#define KT_STRIDE 66   // padded __bf16 stride per d-row (64 + 2) -> conflict-free
#define QPB 4          // queries per block; 2 waves per query -> 8 waves / 256 threads
#define NCHUNK (SN / CHUNK)
#define LDIT ((CHUNK * DN) / (256 * 4))   // float4 staging iterations per chunk

typedef __attribute__((ext_vector_type(16))) __bf16 bf16x16;
typedef __attribute__((ext_vector_type(8)))  float  f32x8;

__device__ __forceinline__ float bf16lo_f32(unsigned u) {
  union { unsigned u; float f; } x; x.u = u << 16; return x.f;
}
__device__ __forceinline__ float bf16hi_f32(unsigned u) {
  union { unsigned u; float f; } x; x.u = u & 0xFFFF0000u; return x.f;
}
__device__ __forceinline__ bf16x16 frag_cast(const unsigned* u) {
  bf16x16 r; __builtin_memcpy(&r, u, 32); return r;
}
__device__ __forceinline__ f32x8 vzero8() {
  f32x8 z;
  #pragma unroll
  for (int i = 0; i < 8; ++i) z[i] = 0.0f;
  return z;
}

__global__ __launch_bounds__(256) void attn_jac_kernel(
    const float* __restrict__ Qm, const float* __restrict__ Km,
    const float* __restrict__ Vm, float* __restrict__ Out)
{
  __shared__ __bf16 ktS[DN * KT_STRIDE];   // K chunk, bf16, [d][s] padded
  __shared__ __bf16 vtS[DN * KT_STRIDE];   // V chunk, bf16, [d][s] padded
  __shared__ float  scoresS[QPB * SN];     // per-query attn weights
  __shared__ float  qS[QPB * DN];
  __shared__ float  wvS[QPB * DN];
  __shared__ float  wkS[QPB * DN];

  const int tid  = threadIdx.x;
  const int wave = tid >> 5;      // 0..7
  const int lane = tid & 31;
  const int qIdx = wave >> 1;     // 0..3 : query within block
  const int h    = wave & 1;      // which half of this query's work
  const int b    = blockIdx.y;
  const int q    = blockIdx.x * QPB + qIdx;

  const float* Kb = Km + (size_t)b * SN * DN;
  const float* Vb = Vm + (size_t)b * SN * DN;

  // ---- load Q tile (4 queries x 64) : one element per thread ----
  qS[tid] = Qm[((size_t)(b * QN + blockIdx.x * QPB + (tid >> 6))) * DN + (tid & 63)];
  __syncthreads();

  float* scoresW = scoresS + qIdx * SN;
  const float smScale = 0.125f;  // 1/sqrt(64)

  // ================= Phase A: scores = scale * Q.K^T (pipelined) =================
  float4 pk[LDIT], pv[LDIT];
  #pragma unroll
  for (int t = 0; t < LDIT; ++t) pk[t] = *(const float4*)(Kb + (t * 256 + tid) * 4);

  #pragma unroll 1
  for (int c = 0; c < NCHUNK; ++c) {
    // commit prefetched chunk c to LDS (bf16, transposed)
    #pragma unroll
    for (int t = 0; t < LDIT; ++t) {
      int idx = (t * 256 + tid) * 4;
      int s = idx >> 6, d = idx & 63;
      ktS[(d + 0) * KT_STRIDE + s] = (__bf16)pk[t].x;
      ktS[(d + 1) * KT_STRIDE + s] = (__bf16)pk[t].y;
      ktS[(d + 2) * KT_STRIDE + s] = (__bf16)pk[t].z;
      ktS[(d + 3) * KT_STRIDE + s] = (__bf16)pk[t].w;
    }
    __syncthreads();
    // issue prefetch for chunk c+1 (hidden behind score compute)
    {
      int cn = (c + 1 < NCHUNK) ? (c + 1) : 0;
      const float* Ksrc = Kb + (size_t)cn * CHUNK * DN;
      #pragma unroll
      for (int t = 0; t < LDIT; ++t) pk[t] = *(const float4*)(Ksrc + (t * 256 + tid) * 4);
    }
    if ((c & 1) == h) {   // wave-uniform: this wave scores this chunk for its q
      const unsigned* ktU = (const unsigned*)ktS;
      float acc0 = 0.f, acc1 = 0.f;
      #pragma unroll 8
      for (int d = 0; d < DN; ++d) {
        float qd = qS[qIdx * DN + d];
        unsigned kk = ktU[d * (KT_STRIDE / 2) + lane];  // pair (2*lane, 2*lane+1)
        acc0 += qd * bf16lo_f32(kk);
        acc1 += qd * bf16hi_f32(kk);
      }
      scoresW[c * CHUNK + 2 * lane]     = acc0 * smScale;
      scoresW[c * CHUNK + 2 * lane + 1] = acc1 * smScale;
    }
    __syncthreads();
  }

  // ---- issue phase-B chunk-0 prefetch; softmax hides the latency ----
  #pragma unroll
  for (int t = 0; t < LDIT; ++t) {
    pk[t] = *(const float4*)(Kb + (t * 256 + tid) * 4);
    pv[t] = *(const float4*)(Vb + (t * 256 + tid) * 4);
  }

  // ================= softmax (h==0 wave of each pair) =================
  if (h == 0) {
    float mx = -1e30f;
    for (int t = lane; t < SN; t += 32) mx = fmaxf(mx, scoresW[t]);
    #pragma unroll
    for (int off = 16; off > 0; off >>= 1) mx = fmaxf(mx, __shfl_xor(mx, off, 32));
    float sum = 0.f;
    for (int t = lane; t < SN; t += 32) {
      float p = __expf(scoresW[t] - mx);
      scoresW[t] = p;
      sum += p;
    }
    #pragma unroll
    for (int off = 16; off > 0; off >>= 1) sum += __shfl_xor(sum, off, 32);
    float inv = 1.0f / sum;
    for (int t = lane; t < SN; t += 32) scoresW[t] *= inv;
  }
  __syncthreads();

  // ================= Phase B: term1 = (a.V)^T K via WMMA (pipelined) =================
  // This wave owns row-tiles ti = 2h, 2h+1 (32 rows of the 64x64 Jacobian).
  f32x8 accT[2][4];
  f32x8 accK[4];              // attn-broadcast x K -> wk (owned by h==1 wave)
  float wvacc[2];
  #pragma unroll
  for (int t2 = 0; t2 < 2; ++t2) {
    wvacc[t2] = 0.f;
    #pragma unroll
    for (int tj = 0; tj < 4; ++tj) accT[t2][tj] = vzero8();
  }
  #pragma unroll
  for (int tj = 0; tj < 4; ++tj) accK[tj] = vzero8();

  const int M  = lane & 15;   // A: row index; B: column index (N)
  const int lh = lane >> 4;   // lane half selects K sub-range of fragments

  #pragma unroll 1
  for (int c = 0; c < NCHUNK; ++c) {
    // commit prefetched chunk c to LDS
    #pragma unroll
    for (int t = 0; t < LDIT; ++t) {
      int idx = (t * 256 + tid) * 4;
      int s = idx >> 6, d = idx & 63;
      ktS[(d + 0) * KT_STRIDE + s] = (__bf16)pk[t].x;
      ktS[(d + 1) * KT_STRIDE + s] = (__bf16)pk[t].y;
      ktS[(d + 2) * KT_STRIDE + s] = (__bf16)pk[t].z;
      ktS[(d + 3) * KT_STRIDE + s] = (__bf16)pk[t].w;
      vtS[(d + 0) * KT_STRIDE + s] = (__bf16)pv[t].x;
      vtS[(d + 1) * KT_STRIDE + s] = (__bf16)pv[t].y;
      vtS[(d + 2) * KT_STRIDE + s] = (__bf16)pv[t].z;
      vtS[(d + 3) * KT_STRIDE + s] = (__bf16)pv[t].w;
    }
    __syncthreads();
    // issue prefetch for chunk c+1 (hidden behind the WMMA work below)
    {
      int cn = (c + 1 < NCHUNK) ? (c + 1) : 0;
      const float* Ksrc = Kb + (size_t)cn * CHUNK * DN;
      const float* Vsrc = Vb + (size_t)cn * CHUNK * DN;
      #pragma unroll
      for (int t = 0; t < LDIT; ++t) {
        pk[t] = *(const float4*)(Ksrc + (t * 256 + tid) * 4);
        pv[t] = *(const float4*)(Vsrc + (t * 256 + tid) * 4);
      }
    }
    const unsigned* ktU = (const unsigned*)ktS;
    const unsigned* vtU = (const unsigned*)vtS;

    #pragma unroll 1
    for (int ks = 0; ks < CHUNK; ks += 32) {
      bf16x16 aF[2];
      bf16x16 atF;
      #pragma unroll
      for (int v = 0; v < 8; ++v) {
        // A-matrix (16x32 bf16) lane/VGPR k-map
        int kA = ks + ((v < 4) ? (lh * 8 + 2 * v) : (16 + lh * 8 + 2 * (v - 4)));
        float a0 = scoresW[c * CHUNK + kA];
        float a1 = scoresW[c * CHUNK + kA + 1];
        atF[2 * v]     = (__bf16)a0;          // sinks into the h==1 path
        atF[2 * v + 1] = (__bf16)a1;
        int pA = kA >> 1;
        #pragma unroll
        for (int t2 = 0; t2 < 2; ++t2) {
          int i = (h * 2 + t2) * 16 + M;
          unsigned vv = vtU[i * (KT_STRIDE / 2) + pA];  // V[s][i],V[s+1][i] packed
          float p0 = a0 * bf16lo_f32(vv);
          float p1 = a1 * bf16hi_f32(vv);
          wvacc[t2] += p0 + p1;              // wv falls out of A-fragment products
          aF[t2][2 * v]     = (__bf16)p0;
          aF[t2][2 * v + 1] = (__bf16)p1;
        }
      }
      // B fragments loaded one at a time (8 VGPRs live, not 32) and consumed
      // immediately; single wave-uniform branch keeps WMMA in EXEC=all-ones code.
      if (h == 0) {
        #pragma unroll
        for (int tj = 0; tj < 4; ++tj) {
          unsigned bU[8];
          #pragma unroll
          for (int v = 0; v < 8; ++v) {
            int pB = (ks + ((v < 4) ? (lh * 8 + 2 * v) : (16 + lh * 8 + 2 * (v - 4)))) >> 1;
            bU[v] = ktU[(tj * 16 + M) * (KT_STRIDE / 2) + pB];
          }
          bf16x16 bF = frag_cast(bU);
          accT[0][tj] = __builtin_amdgcn_wmma_f32_16x16x32_bf16(
              false, aF[0], false, bF, (short)0, accT[0][tj], false, false);
          accT[1][tj] = __builtin_amdgcn_wmma_f32_16x16x32_bf16(
              false, aF[1], false, bF, (short)0, accT[1][tj], false, false);
        }
      } else {
        #pragma unroll
        for (int tj = 0; tj < 4; ++tj) {
          unsigned bU[8];
          #pragma unroll
          for (int v = 0; v < 8; ++v) {
            int pB = (ks + ((v < 4) ? (lh * 8 + 2 * v) : (16 + lh * 8 + 2 * (v - 4)))) >> 1;
            bU[v] = ktU[(tj * 16 + M) * (KT_STRIDE / 2) + pB];
          }
          bf16x16 bF = frag_cast(bU);
          accT[0][tj] = __builtin_amdgcn_wmma_f32_16x16x32_bf16(
              false, aF[0], false, bF, (short)0, accT[0][tj], false, false);
          accT[1][tj] = __builtin_amdgcn_wmma_f32_16x16x32_bf16(
              false, aF[1], false, bF, (short)0, accT[1][tj], false, false);
          accK[tj] = __builtin_amdgcn_wmma_f32_16x16x32_bf16(
              false, atF, false, bF, (short)0, accK[tj], false, false);
        }
      }
    }
    __syncthreads();
  }

  // ================= epilogue: out = scale*(term1 - wv wk^T) =================
  #pragma unroll
  for (int t2 = 0; t2 < 2; ++t2) {
    float w = wvacc[t2] + __shfl_xor(wvacc[t2], 16, 32);  // combine k-halves
    if (lane < 16) wvS[qIdx * DN + (h * 2 + t2) * 16 + lane] = w;
  }
  if (h == 1) {
    #pragma unroll
    for (int tj = 0; tj < 4; ++tj) {
      if (lane < 16) wkS[qIdx * DN + tj * 16 + lane] = accK[tj][0];  // row M=0
    }
  }
  __syncthreads();

  float* outQ = Out + ((size_t)(b * QN + q)) * DN * DN;
  #pragma unroll
  for (int t2 = 0; t2 < 2; ++t2) {
    int ti = h * 2 + t2;
    #pragma unroll
    for (int tj = 0; tj < 4; ++tj) {
      float wkj = wkS[qIdx * DN + tj * 16 + M];
      #pragma unroll
      for (int r = 0; r < 8; ++r) {
        int i = ti * 16 + r + lh * 8;     // C/D layout: M = r + 8*(lane>=16)
        int j = tj * 16 + M;              // N = lane & 15
        float wvi = wvS[qIdx * DN + i];
        outQ[(size_t)i * DN + j] = smScale * (accT[t2][tj][r] - wvi * wkj);
      }
    }
  }
}

extern "C" void kernel_launch(void* const* d_in, const int* in_sizes, int n_in,
                              void* d_out, int out_size, void* d_ws, size_t ws_size,
                              hipStream_t stream) {
  (void)in_sizes; (void)n_in; (void)out_size; (void)d_ws; (void)ws_size;
  const float* q = (const float*)d_in[0];
  const float* k = (const float*)d_in[1];
  const float* v = (const float*)d_in[2];
  float* out = (float*)d_out;
  dim3 grid(QN / QPB, BQ);
  dim3 block(256);
  attn_jac_kernel<<<grid, block, 0, stream>>>(q, k, v, out);
}
